// CausalSelfAttentionRoPE_33088428048910
// MI455X (gfx1250) — matrix operand order repaired and checked by reading, not compiled, not run on hardware
//
#include <hip/hip_runtime.h>

// ---------------------------------------------------------------------------
// CDNA5 (gfx1250, wave32) fused causal self-attention with RoPE.
// All matmul stages run on v_wmma_f32_16x16x32_bf16 (f32 accumulate).
// GEMM K-loops are software-pipelined (double-buffered register fragments),
// unrolled x2 so the buffer rotation is done by register renaming (no movs).
// ---------------------------------------------------------------------------

typedef __bf16 bf16;
typedef __attribute__((ext_vector_type(16))) __bf16 bf16x16;
typedef __attribute__((ext_vector_type(8)))  __bf16 bf16x8;
typedef __attribute__((ext_vector_type(4)))  __bf16 bf16x4;
typedef __attribute__((ext_vector_type(8)))  float  f32x8;

#define D_MODEL 1024
#define N_HEADS 16
#define HEAD_DIM 64
#define BATCH 2
#define SEQ 2048
#define NTOK (BATCH * SEQ)      // 4096 tokens
#define C3 (3 * D_MODEL)        // 3072

static __device__ inline f32x8 zero8() {
  f32x8 z;
#pragma unroll
  for (int i = 0; i < 8; ++i) z[i] = 0.f;
  return z;
}

// A-matrix fragment (16x32 bf16, MxK), source row-major [row][k], leading dim ld.
// Lane layout: lanes 0-15 row=lane, K k0..k0+15 contiguous;
// lanes 16-31 row=lane-16, K k0+16..k0+31.
static __device__ inline bf16x16 load_a16(const bf16* base, int ld, int row_base, int k0) {
  const int lane = threadIdx.x & 31;
  const int grp  = lane >> 4;
  const bf16* p  = base + (size_t)(row_base + (lane & 15)) * ld + k0 + grp * 16;
  bf16x8 lo = *(const bf16x8*)(p);
  bf16x8 hi = *(const bf16x8*)(p + 8);
  bf16x16 a;
#pragma unroll
  for (int i = 0; i < 8; ++i) { a[i] = lo[i]; a[i + 8] = hi[i]; }
  return a;
}

// B-matrix fragment (32x16 bf16, KxN) where memory holds src[n][k] row-major
// (i.e. B[k][n] = src[n*ld + k]).  Lanes 0-15: col n, K k0..k0+15 contiguous;
// lanes 16-31: col n, K k0+16..k0+31.
static __device__ inline bf16x16 load_b16(const bf16* base, int ld, int n_base, int k0) {
  const int lane = threadIdx.x & 31;
  const int grp  = lane >> 4;
  const bf16* p  = base + (size_t)(n_base + (lane & 15)) * ld + k0 + grp * 16;
  bf16x8 lo = *(const bf16x8*)(p);
  bf16x8 hi = *(const bf16x8*)(p + 8);
  bf16x16 b;
#pragma unroll
  for (int i = 0; i < 8; ++i) { b[i] = lo[i]; b[i + 8] = hi[i]; }
  return b;
}

static __device__ inline f32x8 wmma_bf16(bf16x16 a, bf16x16 b, f32x8 c) {
  return __builtin_amdgcn_wmma_f32_16x16x32_bf16(false, a, false, b, (short)0, c,
                                                 false, false);
}

// ---------------------------------------------------------------------------
// K0: fp32 -> bf16 conversion, 4 elements/thread (b128 in, b64 out)
// ---------------------------------------------------------------------------
__global__ void cvt_bf16(const float* __restrict__ in, bf16* __restrict__ out, int n4) {
  int i = blockIdx.x * blockDim.x + threadIdx.x;
  if (i < n4) {
    float4 v = ((const float4*)in)[i];
    bf16x4 o;
    o[0] = (bf16)v.x; o[1] = (bf16)v.y; o[2] = (bf16)v.z; o[3] = (bf16)v.w;
    ((bf16x4*)out)[i] = o;
  }
}

// ---------------------------------------------------------------------------
// K1: qkv = x @ qkv_w^T + b, fused RoPE, scatter into Q/K [BH][T][64] and
//     V transposed [BH][64][T] (so PV B-fragments are contiguous loads).
// Block = 128 threads (4 waves); block tile 128 rows x 64 cols,
// wave tile 32 rows x 64 cols (8 accumulators, 8 WMMA per k-step).
// ---------------------------------------------------------------------------
__global__ void __launch_bounds__(128)
qkv_rope_kernel(const bf16* __restrict__ Xb, const bf16* __restrict__ Wq,
                const float* __restrict__ bias,
                bf16* __restrict__ Qo, bf16* __restrict__ Ko, bf16* __restrict__ Vt) {
  __shared__ __align__(16) float tile[128][65];
  const int col0  = blockIdx.x * 64;
  const int row0  = blockIdx.y * 128;
  const int wave  = threadIdx.x >> 5;
  const int lane  = threadIdx.x & 31;
  const int rbase = row0 + wave * 32;

  f32x8 acc[2][4];
#pragma unroll
  for (int i = 0; i < 2; ++i)
#pragma unroll
    for (int j = 0; j < 4; ++j) acc[i][j] = zero8();

  // ---- software-pipelined K loop (double-buffered fragments, unroll x2) --
  bf16x16 a0 = load_a16(Xb, D_MODEL, rbase, 0);
  bf16x16 a1 = load_a16(Xb, D_MODEL, rbase + 16, 0);
  bf16x16 bc[4];
#pragma unroll
  for (int j = 0; j < 4; ++j) bc[j] = load_b16(Wq, D_MODEL, col0 + j * 16, 0);

#pragma unroll 2
  for (int k0 = 0; k0 < D_MODEL - 32; k0 += 32) {
    __builtin_prefetch((const void*)(Xb + (size_t)rbase * D_MODEL + k0 + 64), 0, 3);
    bf16x16 an0 = load_a16(Xb, D_MODEL, rbase, k0 + 32);
    bf16x16 an1 = load_a16(Xb, D_MODEL, rbase + 16, k0 + 32);
    bf16x16 bn[4];
#pragma unroll
    for (int j = 0; j < 4; ++j) bn[j] = load_b16(Wq, D_MODEL, col0 + j * 16, k0 + 32);
#pragma unroll
    for (int j = 0; j < 4; ++j) {
      acc[0][j] = wmma_bf16(a0, bc[j], acc[0][j]);
      acc[1][j] = wmma_bf16(a1, bc[j], acc[1][j]);
    }
    a0 = an0; a1 = an1;
#pragma unroll
    for (int j = 0; j < 4; ++j) bc[j] = bn[j];
  }
#pragma unroll
  for (int j = 0; j < 4; ++j) {          // drain
    acc[0][j] = wmma_bf16(a0, bc[j], acc[0][j]);
    acc[1][j] = wmma_bf16(a1, bc[j], acc[1][j]);
  }

  // Spill C-layout accumulators to LDS so the RoPE epilogue can read pairs.
  {
    const int rr = wave * 32 + (lane >> 4) * 8;
    const int cc = lane & 15;
#pragma unroll
    for (int j = 0; j < 4; ++j)
#pragma unroll
      for (int r = 0; r < 8; ++r) {
        tile[rr + r][j * 16 + cc]      = acc[0][j][r];
        tile[rr + 16 + r][j * 16 + cc] = acc[1][j][r];
      }
  }
  __syncthreads();

  for (int idx = threadIdx.x; idx < 128 * 64; idx += 128) {
    const int r  = idx >> 6;
    const int cc = idx & 63;
    const int m  = col0 + cc;              // column in [0, 3C)
    const int n  = row0 + r;               // token in [0, NTOK)
    float v = tile[r][cc] + bias[m];
    const int sec = m >> 10;               // 0=q 1=k 2=v
    const int cp  = m & 1023;
    const int h   = cp >> 6;
    const int d   = cp & 63;
    const int b   = n >> 11;
    const int t   = n & 2047;
    if (sec < 2) {
      // RoPE: pairs (2i, 2i+1); rot[2i] = -x[2i+1], rot[2i+1] = x[2i]
      const float vp  = tile[r][cc ^ 1] + bias[m ^ 1];
      const float rot = (d & 1) ? vp : -vp;
      const int   j2  = d & 31;
      const float inv = __powf(10000.f, -(float)(2 * j2) * (1.f / 64.f));
      const float ang = (float)t * inv;
      const float vr  = v * __cosf(ang) + rot * __sinf(ang);
      bf16* dst = (sec == 0) ? Qo : Ko;
      dst[(((size_t)(b * N_HEADS + h) * SEQ) + t) * HEAD_DIM + d] = (bf16)vr;
    } else {
      Vt[(((size_t)(b * N_HEADS + h) * HEAD_DIM) + d) * SEQ + t] = (bf16)v;
    }
  }
}

// ---------------------------------------------------------------------------
// K2: flash-style causal attention per (b, h, 64-query tile).
// Block = 128 threads (4 waves); each wave owns 16 query rows.
// V fragments are loaded at the top of each key-tile iteration so the loads
// overlap the score WMMAs + softmax VALU.
// ---------------------------------------------------------------------------
__global__ void __launch_bounds__(128)
attn_kernel(const bf16* __restrict__ Q, const bf16* __restrict__ K,
            const bf16* __restrict__ Vt, const int* __restrict__ amask,
            bf16* __restrict__ O) {
  __shared__ __align__(16) bf16 pbuf[4][16][64];   // per-wave P staging (8 KB)
  const int bh  = blockIdx.x >> 5;                  // b*16 + h
  const int q0  = (blockIdx.x & 31) * 64;
  const int b   = bh >> 4;
  const int h   = bh & 15;
  const int wave = threadIdx.x >> 5;
  const int lane = threadIdx.x & 31;
  const int grp  = lane >> 4;

  const bf16* Qh  = Q  + (size_t)bh * SEQ * HEAD_DIM;
  const bf16* Kh  = K  + (size_t)bh * SEQ * HEAD_DIM;
  const bf16* Vth = Vt + (size_t)bh * HEAD_DIM * SEQ;

  const int qr = q0 + wave * 16;
  const bf16x16 aq0 = load_a16(Qh, HEAD_DIM, qr, 0);
  const bf16x16 aq1 = load_a16(Qh, HEAD_DIM, qr, 32);

  f32x8 oacc[4];
#pragma unroll
  for (int j = 0; j < 4; ++j) oacc[j] = zero8();
  float mrun[8], lrun[8];
#pragma unroll
  for (int r = 0; r < 8; ++r) { mrun[r] = -1e30f; lrun[r] = 0.f; }

  for (int kk0 = 0; kk0 <= q0; kk0 += 64) {
    // ---- V fragments (independent of softmax): issue loads early --------
    bf16x16 bv0[4], bv1[4];
#pragma unroll
    for (int j = 0; j < 4; ++j) {
      bv0[j] = load_b16(Vth, SEQ, j * 16, kk0);
      bv1[j] = load_b16(Vth, SEQ, j * 16, kk0 + 32);
    }
    // ---- scores S = (Q Kt) / sqrt(Dh), 16x64 per wave -------------------
    f32x8 s[4];
#pragma unroll
    for (int j = 0; j < 4; ++j) {
      bf16x16 bk0 = load_b16(Kh, HEAD_DIM, kk0 + j * 16, 0);
      bf16x16 bk1 = load_b16(Kh, HEAD_DIM, kk0 + j * 16, 32);
      s[j] = wmma_bf16(aq0, bk0, zero8());
      s[j] = wmma_bf16(aq1, bk1, s[j]);
    }
    const bool diag = (kk0 == q0);   // only the diagonal tile needs causal mask
#pragma unroll
    for (int j = 0; j < 4; ++j) {
      const int kj    = kk0 + j * 16 + (lane & 15);
      const int keyok = amask[b * SEQ + kj];
#pragma unroll
      for (int r = 0; r < 8; ++r) {
        float sv = s[j][r] * 0.125f;          // 1/sqrt(64)
        const int qi = qr + grp * 8 + r;
        if (!keyok || (diag && kj > qi)) sv = -1e30f;
        s[j][r] = sv;
      }
    }
    // ---- online softmax (row reductions across the 16-lane col group) ---
#pragma unroll
    for (int r = 0; r < 8; ++r) {
      float v = fmaxf(fmaxf(s[0][r], s[1][r]), fmaxf(s[2][r], s[3][r]));
#pragma unroll
      for (int mm = 1; mm < 16; mm <<= 1) v = fmaxf(v, __shfl_xor(v, mm, 32));
      const float mn = fmaxf(mrun[r], v);
      const float sc = __expf(mrun[r] - mn);
      mrun[r] = mn;
      float rs = 0.f;
#pragma unroll
      for (int j = 0; j < 4; ++j) {
        const float p = __expf(s[j][r] - mn);
        s[j][r] = p;
        rs += p;
      }
#pragma unroll
      for (int mm = 1; mm < 16; mm <<= 1) rs += __shfl_xor(rs, mm, 32);
      lrun[r] = lrun[r] * sc + rs;
#pragma unroll
      for (int j = 0; j < 4; ++j) oacc[j][r] *= sc;
    }
    // ---- P: C-layout -> A-layout through LDS, then O += P @ V -----------
#pragma unroll
    for (int j = 0; j < 4; ++j)
#pragma unroll
      for (int r = 0; r < 8; ++r)
        pbuf[wave][grp * 8 + r][j * 16 + (lane & 15)] = (bf16)s[j][r];
    __syncthreads();
    const bf16* pw = &pbuf[wave][0][0];
    bf16x16 ap0 = load_a16(pw, 64, 0, 0);
    bf16x16 ap1 = load_a16(pw, 64, 0, 32);
#pragma unroll
    for (int j = 0; j < 4; ++j) {
      oacc[j] = wmma_bf16(ap0, bv0[j], oacc[j]);
      oacc[j] = wmma_bf16(ap1, bv1[j], oacc[j]);
    }
    __syncthreads();
  }
  // ---- normalize, query mask, write token-major bf16 --------------------
#pragma unroll
  for (int r = 0; r < 8; ++r) {
    const int   t     = qr + grp * 8 + r;
    const float inv_l = 1.f / fmaxf(lrun[r], 1e-20f);
    const float qm    = (float)amask[b * SEQ + t];
#pragma unroll
    for (int j = 0; j < 4; ++j) {
      const float y = oacc[j][r] * inv_l * qm;
      O[((size_t)(b * SEQ) + t) * D_MODEL + h * HEAD_DIM + j * 16 + (lane & 15)] =
          (bf16)y;
    }
  }
}

// ---------------------------------------------------------------------------
// K3: y = attn_out @ out_w^T + out_b  (fp32 output)
// Block = 128 threads (4 waves); wave tile 32 rows x 64 cols, pipelined x2.
// ---------------------------------------------------------------------------
__global__ void __launch_bounds__(128)
out_gemm_kernel(const bf16* __restrict__ Ob, const bf16* __restrict__ Wo,
                const float* __restrict__ bias, float* __restrict__ Y) {
  const int col0  = blockIdx.x * 64;
  const int row0  = blockIdx.y * 128;
  const int wave  = threadIdx.x >> 5;
  const int lane  = threadIdx.x & 31;
  const int rbase = row0 + wave * 32;

  f32x8 acc[2][4];
#pragma unroll
  for (int i = 0; i < 2; ++i)
#pragma unroll
    for (int j = 0; j < 4; ++j) acc[i][j] = zero8();

  bf16x16 a0 = load_a16(Ob, D_MODEL, rbase, 0);
  bf16x16 a1 = load_a16(Ob, D_MODEL, rbase + 16, 0);
  bf16x16 bc[4];
#pragma unroll
  for (int j = 0; j < 4; ++j) bc[j] = load_b16(Wo, D_MODEL, col0 + j * 16, 0);

#pragma unroll 2
  for (int k0 = 0; k0 < D_MODEL - 32; k0 += 32) {
    __builtin_prefetch((const void*)(Ob + (size_t)rbase * D_MODEL + k0 + 64), 0, 3);
    bf16x16 an0 = load_a16(Ob, D_MODEL, rbase, k0 + 32);
    bf16x16 an1 = load_a16(Ob, D_MODEL, rbase + 16, k0 + 32);
    bf16x16 bn[4];
#pragma unroll
    for (int j = 0; j < 4; ++j) bn[j] = load_b16(Wo, D_MODEL, col0 + j * 16, k0 + 32);
#pragma unroll
    for (int j = 0; j < 4; ++j) {
      acc[0][j] = wmma_bf16(a0, bc[j], acc[0][j]);
      acc[1][j] = wmma_bf16(a1, bc[j], acc[1][j]);
    }
    a0 = an0; a1 = an1;
#pragma unroll
    for (int j = 0; j < 4; ++j) bc[j] = bn[j];
  }
#pragma unroll
  for (int j = 0; j < 4; ++j) {          // drain
    acc[0][j] = wmma_bf16(a0, bc[j], acc[0][j]);
    acc[1][j] = wmma_bf16(a1, bc[j], acc[1][j]);
  }

  const int cc = lane & 15;
  const int rr = (lane >> 4) * 8;
#pragma unroll
  for (int j = 0; j < 4; ++j) {
    const int   m  = col0 + j * 16 + cc;
    const float bv = bias[m];
#pragma unroll
    for (int r = 0; r < 8; ++r) {
      Y[(size_t)(rbase + rr + r) * D_MODEL + m]      = acc[0][j][r] + bv;
      Y[(size_t)(rbase + 16 + rr + r) * D_MODEL + m] = acc[1][j][r] + bv;
    }
  }
}

// ---------------------------------------------------------------------------
// Launch: d_in = {x, attention_mask, qkv_w, qkv_b, out_w, out_b}
// Workspace layout (48 MB total, bf16):
//   [ 0MB) Xb 8MB | [8) Wqkv 6MB | [14) Wout 2MB | [16) Q 8MB |
//   [24) K 8MB    | [32) Vt 8MB  | [40) O 8MB
// ---------------------------------------------------------------------------
extern "C" void kernel_launch(void* const* d_in, const int* in_sizes, int n_in,
                              void* d_out, int out_size, void* d_ws, size_t ws_size,
                              hipStream_t stream) {
  (void)in_sizes; (void)n_in; (void)out_size; (void)ws_size;
  const float* x     = (const float*)d_in[0];
  const int*   amask = (const int*)d_in[1];
  const float* qkv_w = (const float*)d_in[2];
  const float* qkv_b = (const float*)d_in[3];
  const float* out_w = (const float*)d_in[4];
  const float* out_b = (const float*)d_in[5];
  float* y = (float*)d_out;

  char* ws = (char*)d_ws;
  bf16* Xb  = (bf16*)(ws);
  bf16* Wqb = (bf16*)(ws + ((size_t)8 << 20));
  bf16* Wob = (bf16*)(ws + ((size_t)14 << 20));
  bf16* Qb  = (bf16*)(ws + ((size_t)16 << 20));
  bf16* Kb  = (bf16*)(ws + ((size_t)24 << 20));
  bf16* Vtb = (bf16*)(ws + ((size_t)32 << 20));
  bf16* Ob  = (bf16*)(ws + ((size_t)40 << 20));

  cvt_bf16<<<(NTOK * D_MODEL / 4 + 255) / 256, 256, 0, stream>>>(x, Xb,
                                                                 NTOK * D_MODEL / 4);
  cvt_bf16<<<(C3 * D_MODEL / 4 + 255) / 256, 256, 0, stream>>>(qkv_w, Wqb,
                                                               C3 * D_MODEL / 4);
  cvt_bf16<<<(D_MODEL * D_MODEL / 4 + 255) / 256, 256, 0, stream>>>(
      out_w, Wob, D_MODEL * D_MODEL / 4);

  dim3 g1(C3 / 64, NTOK / 128);
  qkv_rope_kernel<<<g1, 128, 0, stream>>>(Xb, Wqb, qkv_b, Qb, Kb, Vtb);

  attn_kernel<<<BATCH * N_HEADS * (SEQ / 64), 128, 0, stream>>>(Qb, Kb, Vtb, amask, Ob);

  dim3 g3(D_MODEL / 64, NTOK / 128);
  out_gemm_kernel<<<g3, 128, 0, stream>>>(Ob, Wob, out_b, y);
}
